// VariationalGCNEncoder_89618787598755
// MI455X (gfx1250) — matrix operand order repaired and checked by reading, not compiled, not run on hardware
//
#include <hip/hip_runtime.h>
#include <hip/hip_bf16.h>

// ---------------------------------------------------------------------------
// Variational GCN encoder for MI455X (gfx1250, wave32, WMMA).
//   sizes: N=100000, E=1600000, IN=256, HID=128, OUT=64
//   layer1: h = relu(prop(x @ W1) + b1)
//   fused : [mu|ls]_pre = h @ [W_mu|W_ls]; one scatter pass; split + bias
// ---------------------------------------------------------------------------

typedef __bf16 bf16_t;
typedef __attribute__((ext_vector_type(16))) __bf16 v16bf;
typedef __attribute__((ext_vector_type(8)))  __bf16 v8bf;
typedef __attribute__((ext_vector_type(8)))  float  v8f;

#define IN_C  256
#define HID_C 128
#define OUT_C 64

// ---------------------------------------------------------------------------
// Elementwise helpers
// ---------------------------------------------------------------------------
__global__ void k_fill1(float* __restrict__ p, int n) {
  int i = blockIdx.x * blockDim.x + threadIdx.x;
  if (i < n) p[i] = 1.0f;                       // self-loop contribution to degree
}

__global__ void k_deg_edges(float* __restrict__ deg, const long long* __restrict__ ei, int E) {
  int e = blockIdx.x * blockDim.x + threadIdx.x;
  if (e < E) atomicAdd(&deg[(int)ei[(size_t)E + e]], 1.0f);   // in-degree over dst
}

__global__ void k_rsqrt(const float* __restrict__ deg, float* __restrict__ dinv, int n) {
  int i = blockIdx.x * blockDim.x + threadIdx.x;
  if (i < n) dinv[i] = rsqrtf(deg[i]);          // deg >= 1 always (self-loop)
}

__global__ void k_cvt_bf16(const float* __restrict__ in, bf16_t* __restrict__ out, size_t n) {
  size_t i = (size_t)blockIdx.x * blockDim.x + threadIdx.x;
  if (i < n) out[i] = (bf16_t)in[i];
}

// out[n*K + k] = (bf16) in[k*NC + n]   (row-major [K,NC] -> transposed bf16 [NC,K])
__global__ void k_transpose_bf16(const float* __restrict__ in, bf16_t* __restrict__ out,
                                 int K, int NC) {
  int idx = blockIdx.x * blockDim.x + threadIdx.x;
  if (idx >= K * NC) return;
  int k = idx / NC;
  int n = idx - k * NC;
  out[(size_t)n * K + k] = (bf16_t)in[(size_t)k * NC + n];
}

// ---------------------------------------------------------------------------
// bf16 WMMA GEMM:  C[M, NTILES*16] = A[M, KTILES*32] (bf16, row-major)
//                  x  B  given as BT[NTILES*16, KTILES*32] (bf16, N-major rows)
// Weight panel staged once per workgroup in LDS (ds_load feeds WMMA B operand,
// decoupled from the A global-load counter). One wave = one 16-row strip.
// Fragment layouts per CDNA5 ISA 7.12.2 (16-bit A 16x32, B 32x16, f32 C/D).
// ---------------------------------------------------------------------------
template <int KTILES, int NTILES>
__global__ __launch_bounds__(128)
void k_gemm_bf16_wmma(const bf16_t* __restrict__ A,
                      const bf16_t* __restrict__ BT,
                      float* __restrict__ C,
                      int numRowTiles) {
  constexpr int K  = KTILES * 32;
  constexpr int NC = NTILES * 16;

  __shared__ __align__(32) bf16_t sB[NC * K];   // 64 KB (GEMM1) / 32 KB (GEMM2)
  {
    constexpr int CH = NC * K * 2 / 16;         // 16B chunks
    const uint4* src = (const uint4*)BT;
    uint4*       dst = (uint4*)sB;
    for (int i = threadIdx.x; i < CH; i += 128) dst[i] = src[i];
  }
  __syncthreads();                              // before any wave exits

  const int lane    = threadIdx.x & 31;
  const int wave    = threadIdx.x >> 5;
  const int rowTile = blockIdx.x * 4 + wave;
  if (rowTile >= numRowTiles) return;           // uniform per wave: EXEC stays all-ones

  const int r  = lane & 15;                     // row-in-tile (A), col (B/C)
  const int hh = lane >> 4;                     // lane half selects K sub-range / M half

  v8f acc[NTILES];
#pragma unroll
  for (int i = 0; i < NTILES; ++i) acc[i] = (v8f){0.f,0.f,0.f,0.f,0.f,0.f,0.f,0.f};

  // A: lane half 0 holds K = {k0..k0+7, k0+16..k0+23}; half 1 holds {+8, +24}
  const bf16_t* aRow = A + ((size_t)rowTile * 16 + r) * K + hh * 8;

#pragma unroll
  for (int kt = 0; kt < KTILES; ++kt) {
    const bf16_t* p = aRow + kt * 32;
    if (kt + 1 < KTILES) __builtin_prefetch(aRow + (kt + 1) * 32, 0, 1);
    v8bf lo = *(const v8bf*)(p);
    v8bf hi = *(const v8bf*)(p + 16);
    v16bf a = __builtin_shufflevector(lo, hi, 0,1,2,3,4,5,6,7,8,9,10,11,12,13,14,15);
#pragma unroll
    for (int nt = 0; nt < NTILES; ++nt) {
      // B: lane half 0 holds K = {k0..k0+7, k0+16..k0+23} of column r; half 1 holds {+8, +24}
      const bf16_t* q = sB + ((size_t)nt * 16 + r) * K + kt * 32 + hh * 8;
      v8bf blo = *(const v8bf*)(q);
      v8bf bhi = *(const v8bf*)(q + 16);
      v16bf b = __builtin_shufflevector(blo, bhi, 0,1,2,3,4,5,6,7,8,9,10,11,12,13,14,15);
      acc[nt] = __builtin_amdgcn_wmma_f32_16x16x32_bf16(
          /*neg_a=*/false, a, /*neg_b=*/false, b,
          /*c_mod=*/(short)0, acc[nt], /*reuse_a=*/false, /*reuse_b=*/false);
    }
  }

  // D layout: VGPR i, lanes 0-15 -> M=i, lanes 16-31 -> M=8+i; N = lane&15
#pragma unroll
  for (int nt = 0; nt < NTILES; ++nt) {
    float* cp = C + ((size_t)rowTile * 16 + hh * 8) * NC + nt * 16 + r;
#pragma unroll
    for (int i = 0; i < 8; ++i) cp[(size_t)i * NC] = acc[nt][i];
  }
}

// ---------------------------------------------------------------------------
// Propagation: agg[i,:] = dinv[i]^2 * pre[i,:]   (self-loop, non-atomic init)
//              agg[d,:] += dinv[s]*dinv[d]*pre[s,:]  over edges (atomics in L2)
// ---------------------------------------------------------------------------
template <int F>
__global__ void k_self_init(const float* __restrict__ pre, const float* __restrict__ dinv,
                            float* __restrict__ agg, int N) {
  size_t idx = (size_t)blockIdx.x * blockDim.x + threadIdx.x;
  if (idx >= (size_t)N * F) return;
  int i = (int)(idx / F);
  float w = dinv[i];
  agg[idx] = w * w * pre[idx];
}

template <int F>
__global__ void k_scatter(const float* __restrict__ pre, float* __restrict__ agg,
                          const float* __restrict__ dinv,
                          const long long* __restrict__ ei, int E) {
  constexpr int PER = F / 32;                   // wave32: floats per lane
  size_t gid = (size_t)blockIdx.x * blockDim.x + threadIdx.x;
  int e    = (int)(gid >> 5);                   // one wave per edge (uniform)
  int lane = (int)(gid & 31);
  if (e >= E) return;
  int s = (int)ei[e];
  int d = (int)ei[(size_t)E + e];
  float w = dinv[s] * dinv[d];
  const float* sp = pre + (size_t)s * F + lane * PER;
  float*       dp = agg + (size_t)d * F + lane * PER;
  float v[PER];
#pragma unroll
  for (int j = 0; j < PER; ++j) v[j] = sp[j];   // coalesces to b128 load
#pragma unroll
  for (int j = 0; j < PER; ++j) atomicAdd(&dp[j], w * v[j]);
}

template <int F>
__global__ void k_relu_bias_bf16(const float* __restrict__ agg, const float* __restrict__ b,
                                 bf16_t* __restrict__ out, int N) {
  size_t idx = (size_t)blockIdx.x * blockDim.x + threadIdx.x;
  if (idx >= (size_t)N * F) return;
  int j = (int)(idx & (F - 1));
  float v = agg[idx] + b[j];
  out[idx] = (bf16_t)(v > 0.f ? v : 0.f);
}

// comb [N,128] = [mu_agg | ls_agg]  ->  d_out layout [N,64] ++ [N,64], plus biases
__global__ void k_split_bias(const float* __restrict__ comb,
                             const float* __restrict__ bmu, const float* __restrict__ bls,
                             float* __restrict__ out_mu, float* __restrict__ out_ls, int N) {
  size_t idx = (size_t)blockIdx.x * blockDim.x + threadIdx.x;
  if (idx >= (size_t)N * OUT_C) return;
  int i = (int)(idx >> 6);
  int j = (int)(idx & 63);
  out_mu[idx] = comb[(size_t)i * 128 + j]      + bmu[j];
  out_ls[idx] = comb[(size_t)i * 128 + 64 + j] + bls[j];
}

// ---------------------------------------------------------------------------
// Host launch
// ---------------------------------------------------------------------------
extern "C" void kernel_launch(void* const* d_in, const int* in_sizes, int n_in,
                              void* d_out, int out_size, void* d_ws, size_t ws_size,
                              hipStream_t stream) {
  (void)n_in; (void)out_size; (void)ws_size;

  const float*     x   = (const float*)d_in[0];
  const long long* ei  = (const long long*)d_in[1];   // int64 [2,E]: src then dst
  const float*     W1  = (const float*)d_in[2];       // [256,128]
  const float*     b1  = (const float*)d_in[3];       // [128]
  const float*     Wmu = (const float*)d_in[4];       // [128,64]
  const float*     bmu = (const float*)d_in[5];       // [64]
  const float*     Wls = (const float*)d_in[6];       // [128,64]
  const float*     bls = (const float*)d_in[7];       // [64]

  const int N = in_sizes[0] / IN_C;                   // 100000
  const int E = in_sizes[1] / 2;                      // 1600000

  // ---- workspace carve (256B aligned regions) ----
  char* ws = (char*)d_ws;
  size_t off = 0;
  auto carve = [&](size_t bytes) -> char* {
    char* p = ws + off;
    off = (off + bytes + 255) & ~(size_t)255;
    return p;
  };
  float*  deg   = (float*) carve((size_t)N * 4);
  float*  dinv  = (float*) carve((size_t)N * 4);
  bf16_t* xb    = (bf16_t*)carve((size_t)N * IN_C * 2);           // 51.2 MB
  bf16_t* W1T   = (bf16_t*)carve((size_t)HID_C * IN_C * 2);       // 64 KB
  bf16_t* WcatT = (bf16_t*)carve((size_t)(2 * OUT_C) * HID_C * 2);// 32 KB [mu rows | ls rows]
  float*  hw    = (float*) carve((size_t)N * HID_C * 4);          // 51.2 MB
  float*  agg1  = (float*) carve((size_t)N * HID_C * 4);          // 51.2 MB
  bf16_t* hb    = (bf16_t*)xb;   // alias: xb dead after GEMM1
  float*  pre2  = hw;            // alias: hw dead after layer-1 scatter
  float*  comb  = agg1;          // alias: agg1 dead after relu/bias -> hb

  float* out_mu = (float*)d_out;                  // [N,64]
  float* out_ls = (float*)d_out + (size_t)N * OUT_C;

  const int TB = 256;
  auto blocks = [](size_t work, int tb) { return (unsigned)((work + tb - 1) / tb); };

  // ---- cached normalization: deg (incl. self-loop) -> deg^-1/2 ----
  k_fill1<<<blocks(N, TB), TB, 0, stream>>>(deg, N);
  k_deg_edges<<<blocks(E, TB), TB, 0, stream>>>(deg, ei, E);
  k_rsqrt<<<blocks(N, TB), TB, 0, stream>>>(deg, dinv, N);

  // ---- precision staging: bf16 activations, transposed bf16 weights ----
  k_cvt_bf16<<<blocks((size_t)N * IN_C, TB), TB, 0, stream>>>(x, xb, (size_t)N * IN_C);
  k_transpose_bf16<<<blocks((size_t)IN_C * HID_C, TB), TB, 0, stream>>>(W1, W1T, IN_C, HID_C);
  // concatenated [W_mu | W_ls] -> WcatT rows 0..63 (mu) and 64..127 (ls)
  k_transpose_bf16<<<blocks((size_t)HID_C * OUT_C, TB), TB, 0, stream>>>(
      Wmu, WcatT, HID_C, OUT_C);
  k_transpose_bf16<<<blocks((size_t)HID_C * OUT_C, TB), TB, 0, stream>>>(
      Wls, WcatT + (size_t)OUT_C * HID_C, HID_C, OUT_C);

  const int rowTiles = N / 16;                        // 6250 (exact)
  const unsigned gemmGrid = (unsigned)((rowTiles + 3) / 4);

  // ---- layer 1: h = relu(prop(x @ W1) + b1) ----
  k_gemm_bf16_wmma<IN_C / 32, HID_C / 16><<<gemmGrid, 128, 0, stream>>>(xb, W1T, hw, rowTiles);
  k_self_init<HID_C><<<blocks((size_t)N * HID_C, TB), TB, 0, stream>>>(hw, dinv, agg1, N);
  k_scatter<HID_C><<<blocks((size_t)E * 32, TB), TB, 0, stream>>>(hw, agg1, dinv, ei, E);
  k_relu_bias_bf16<HID_C><<<blocks((size_t)N * HID_C, TB), TB, 0, stream>>>(agg1, b1, hb, N);

  // ---- fused layer 2/3: [mu|ls] = prop(h @ [W_mu|W_ls]) + [b_mu|b_ls] ----
  k_gemm_bf16_wmma<HID_C / 32, (2 * OUT_C) / 16><<<gemmGrid, 128, 0, stream>>>(
      hb, WcatT, pre2, rowTiles);
  k_self_init<2 * OUT_C><<<blocks((size_t)N * 2 * OUT_C, TB), TB, 0, stream>>>(
      pre2, dinv, comb, N);
  k_scatter<2 * OUT_C><<<blocks((size_t)E * 32, TB), TB, 0, stream>>>(
      pre2, comb, dinv, ei, E);
  k_split_bias<<<blocks((size_t)N * OUT_C, TB), TB, 0, stream>>>(
      comb, bmu, bls, out_mu, out_ls, N);
}
